// HilbertAttentionTritonFixed_23029614641320
// MI455X (gfx1250) — compile-verified, matrix operand order checked by reading, not run on hardware
//
#include <hip/hip_runtime.h>

// ---------------------------------------------------------------------------
// MI455X (gfx1250) Hilbert attention:
//   K1: qkv = x @ Wqkv^T            (bf16 WMMA, fp32 acc, q pre-scaled by 1/8)
//   K2: segmented hilbert attention  (bf16 WMMA for QK^T and W*V)
//   K3: out = attn @ Wout^T          (bf16 WMMA, fp32 out)
// Intermediates live in d_ws as bf16 (96MB qkv + 32MB attn).
// GEMM main loop is software-pipelined: global->reg prefetch overlaps WMMA,
// LDS is double-buffered (one barrier per K-step).
// ---------------------------------------------------------------------------

typedef unsigned short u16;
typedef __attribute__((ext_vector_type(16))) __bf16 v16bf;
typedef __attribute__((ext_vector_type(8)))  float  v8f;
typedef __attribute__((ext_vector_type(4)))  unsigned int u32x4;
typedef __attribute__((ext_vector_type(4)))  float  f32x4;

union Frag { v16bf v; u32x4 q[2]; };

__device__ __forceinline__ u16 f32_to_bf16(float f) {
  unsigned u = __builtin_bit_cast(unsigned, f);
  u += 0x7FFFu + ((u >> 16) & 1u);            // round-to-nearest-even
  return (u16)(u >> 16);
}
__device__ __forceinline__ float bf16_to_f32(u16 h) {
  unsigned u = ((unsigned)h) << 16;
  return __builtin_bit_cast(float, u);
}
__device__ __forceinline__ unsigned pack2(float a, float b) {
  return (unsigned)f32_to_bf16(a) | ((unsigned)f32_to_bf16(b) << 16);
}

__device__ __forceinline__ v8f wmma_bf16(v16bf a, v16bf b, v8f c) {
  // D(f32 16x16) = A(bf16 16x32) * B(bf16 32x16) + C
  return __builtin_amdgcn_wmma_f32_16x16x32_bf16(false, a, false, b,
                                                 (short)0, c, false, false);
}

// Fragment loader for BOTH WMMA operands. Source tile is stored row-major in
// LDS as [16 rows][K] bf16 (A: row = M, B stored as B^T: row = N).
// ISA 16-bit layout: lane<16 -> row=lane,    K in {kk+0..7,  kk+16..23}
//                    lane>=16 -> row=lane-16, K in {kk+8..15, kk+24..31}
__device__ __forceinline__ Frag load_frag(const u16* base, int r0, int kk,
                                          int lane, int ldt) {
  Frag f;
  int lr = lane & 15;
  int kh = (lane >> 4) << 3;                  // 0 or 8
  const u16* p = base + (size_t)(r0 + lr) * ldt + kk + kh;
  f.q[0] = *(const u32x4*)p;                  // 8 bf16 (ds_load_b128)
  f.q[1] = *(const u32x4*)(p + 16);           // 8 bf16
  return f;
}

// ---- register staging for the pipelined GEMM (16 elements per thread) ----
struct StageF32 {
  f32x4 v[4];
  __device__ __forceinline__ void load(const float* p) {
    const f32x4* s = (const f32x4*)p;
    v[0] = s[0]; v[1] = s[1]; v[2] = s[2]; v[3] = s[3];
  }
  __device__ __forceinline__ void store(u16* d) const {
    u32x4 o0, o1;
    o0[0] = pack2(v[0][0], v[0][1]); o0[1] = pack2(v[0][2], v[0][3]);
    o0[2] = pack2(v[1][0], v[1][1]); o0[3] = pack2(v[1][2], v[1][3]);
    o1[0] = pack2(v[2][0], v[2][1]); o1[1] = pack2(v[2][2], v[2][3]);
    o1[2] = pack2(v[3][0], v[3][1]); o1[3] = pack2(v[3][2], v[3][3]);
    ((u32x4*)d)[0] = o0;
    ((u32x4*)d)[1] = o1;
  }
};
struct StageBF16 {
  u32x4 v[2];
  __device__ __forceinline__ void load(const u16* p) {
    const u32x4* s = (const u32x4*)p;
    v[0] = s[0]; v[1] = s[1];
  }
  __device__ __forceinline__ void store(u16* d) const {
    ((u32x4*)d)[0] = v[0];
    ((u32x4*)d)[1] = v[1];
  }
};
template <typename T> struct StageFor;
template <> struct StageFor<float> { using type = StageF32; };
template <> struct StageFor<u16>   { using type = StageBF16; };

__device__ __forceinline__ void store_c(u16* p, float v, bool scale) {
  if (scale) v *= 0.125f;                     // SCALE = HD^-0.5 = 1/8 (exact)
  *p = f32_to_bf16(v);
}
__device__ __forceinline__ void store_c(float* p, float v, bool) { *p = v; }

// ---------------------------------------------------------------------------
// C[MT x N] = A[MT x K] * B[N x K]^T     (A f32 or bf16; B f32; C bf16 or f32)
// 128x128 tile / workgroup, BK=32, 8 waves each computing a 32x64 region
// as 2x4 tiles of v_wmma_f32_16x16x32_bf16. Double-buffered + pipelined.
// ---------------------------------------------------------------------------
template <typename AT, typename CT>
__global__ __launch_bounds__(256) void gemm_bt_kernel(
    const AT* __restrict__ A, const float* __restrict__ B, CT* __restrict__ C,
    int N, int K, int scale_cols)
{
  constexpr int BM = 128, BN = 128, BK = 32, LDT = 40;   // LDT*2 = 80B stride
  __shared__ __align__(16) u16 As[2][BM * LDT];
  __shared__ __align__(16) u16 Bs[2][BN * LDT];

  const int tid  = threadIdx.x;
  const int lane = tid & 31;
  const int wave = tid >> 5;
  const int wr = wave & 3;        // 4 row groups of 32
  const int wc = wave >> 2;       // 2 col groups of 64
  const int mBase = blockIdx.y * BM;
  const int nBase = blockIdx.x * BN;

  const int lrow = tid >> 1;            // 0..127
  const int lcol = (tid & 1) << 4;      // 0 or 16
  const int soff = lrow * LDT + lcol;   // LDS staging offset (u16 units)

  v8f acc[2][4];
  v8f vzero = {};
#pragma unroll
  for (int mi = 0; mi < 2; ++mi)
#pragma unroll
    for (int ni = 0; ni < 4; ++ni) acc[mi][ni] = vzero;

  const AT*    aSrc = A + (size_t)(mBase + lrow) * K + lcol;
  const float* bSrc = B + (size_t)(nBase + lrow) * K + lcol;

  typename StageFor<AT>::type aReg;
  StageF32 bReg;

  // prologue: stage slab 0 into LDS buffer 0
  aReg.load(aSrc);
  bReg.load(bSrc);
  aReg.store(&As[0][soff]);
  bReg.store(&Bs[0][soff]);
  __syncthreads();

  int buf = 0;
  for (int k0 = 0; k0 < K; k0 += BK) {
    const bool more = (k0 + BK) < K;
    if (more) {                                   // global->reg for next slab
      aReg.load(aSrc + k0 + BK);
      bReg.load(bSrc + k0 + BK);
    }
    if (k0 + 2 * BK < K) {                        // pull slab k+2 toward L2
      __builtin_prefetch(aSrc + k0 + 2 * BK, 0, 1);
      __builtin_prefetch(bSrc + k0 + 2 * BK, 0, 1);
    }

    const u16* Ab = As[buf];
    const u16* Bb = Bs[buf];
    Frag af0 = load_frag(Ab, wr * 32,      0, lane, LDT);
    Frag af1 = load_frag(Ab, wr * 32 + 16, 0, lane, LDT);
#pragma unroll
    for (int ni = 0; ni < 4; ++ni) {
      Frag bf = load_frag(Bb, wc * 64 + ni * 16, 0, lane, LDT);
      acc[0][ni] = wmma_bf16(af0.v, bf.v, acc[0][ni]);
      acc[1][ni] = wmma_bf16(af1.v, bf.v, acc[1][ni]);
    }

    if (more) {                                   // reg->LDS (other buffer)
      aReg.store(&As[buf ^ 1][soff]);
      bReg.store(&Bs[buf ^ 1][soff]);
    }
    __syncthreads();                              // one barrier per K-step
    buf ^= 1;
  }

  // Epilogue. C/D layout: VGPR r: lanes 0-15 -> (row r0+r, col n0+lane),
  // lanes 16-31 -> (row r0+r+8, col n0+lane-16).
  const int rb = mBase + wr * 32 + ((lane >> 4) << 3);
  const int cb = nBase + wc * 64 + (lane & 15);
#pragma unroll
  for (int mi = 0; mi < 2; ++mi)
#pragma unroll
    for (int ni = 0; ni < 4; ++ni) {
      const int col = cb + ni * 16;
      const bool sc = col < scale_cols;
#pragma unroll
      for (int r = 0; r < 8; ++r) {
        const int row = rb + mi * 16 + r;
        store_c(C + (size_t)row * N + col, acc[mi][ni][r], sc);
      }
    }
}

// ---------------------------------------------------------------------------
// Segmented Hilbert attention. One workgroup per (segment s, head h, batch b).
// qkv buffer layout (bf16): [b][m][comp(3)][h(16)][hd(64)]  (q already *1/8)
// attn output   (bf16): [b*M + i][h*64 + hd]
// ---------------------------------------------------------------------------
__global__ __launch_bounds__(256) void hilbert_attn_kernel(
    const u16* __restrict__ qkv, const int* __restrict__ hmap,
    u16* __restrict__ attn)
{
  constexpr int M = 4096, SEG = 128, T = 64, HD = 64, LDT = 72; // 144B stride
  const int s = blockIdx.x, h = blockIdx.y, b = blockIdx.z;

  __shared__ __align__(16) u16 Qs[SEG * LDT];   // queries [128][64]
  __shared__ __align__(16) u16 Ks[T * LDT];     // keys    [64][64]
  __shared__ __align__(16) u16 Vt[HD * LDT];    // V^T     [hd][key]
  __shared__ __align__(16) u16 Ws[SEG * LDT];   // exp weights [128][64]
  __shared__ float wmax[8][T];                  // per-wave per-key maxima
  __shared__ float den[SEG];

  const int tid = threadIdx.x;
  const int lane = tid & 31;
  const int wave = tid >> 5;

  // ---- gather Q (all 256 threads: 128 rows x 2 halves of 32 bf16) ----
  {
    const int row  = tid >> 1;
    const int half = tid & 1;
    const int g = hmap[s * SEG + row];
    const u32x4* src = (const u32x4*)(qkv + ((size_t)(b * M + g)) * 3072
                                      + h * 64 + half * 32);
    u32x4* dst = (u32x4*)(Qs + row * LDT + half * 32);
    dst[0] = src[0]; dst[1] = src[1]; dst[2] = src[2]; dst[3] = src[3];
  }
  // ---- gather K (threads 0..127) / V transposed (threads 128..255) ----
  if (tid < 128) {
    const int t = tid >> 1, half = tid & 1;
    const int g = hmap[s * SEG + t * 2];                 // dilation 2
    const u32x4* src = (const u32x4*)(qkv + ((size_t)(b * M + g)) * 3072
                                      + 1024 + h * 64 + half * 32);
    u32x4* dst = (u32x4*)(Ks + t * LDT + half * 32);
    dst[0] = src[0]; dst[1] = src[1]; dst[2] = src[2]; dst[3] = src[3];
  } else {
    const int u = tid - 128;
    const int t = u >> 1, half = u & 1;
    const int g = hmap[s * SEG + t * 2];
    const u32x4* src = (const u32x4*)(qkv + ((size_t)(b * M + g)) * 3072
                                      + 2048 + h * 64 + half * 32);
    u32x4 dd[4] = {src[0], src[1], src[2], src[3]};
#pragma unroll
    for (int i = 0; i < 4; ++i)
#pragma unroll
      for (int e = 0; e < 4; ++e) {
        const unsigned w = dd[i][e];
        const int hd = half * 32 + i * 8 + e * 2;
        Vt[hd * LDT + t]       = (u16)(w & 0xFFFFu);
        Vt[(hd + 1) * LDT + t] = (u16)(w >> 16);
      }
  }
  __syncthreads();

  // ---- scores = Q * K^T : wave w owns rows [16w,16w+16), all 64 keys ----
  const int r0 = wave * 16;
  v8f sacc[4];
  v8f vzero = {};
#pragma unroll
  for (int ct = 0; ct < 4; ++ct) sacc[ct] = vzero;
#pragma unroll
  for (int kk = 0; kk < HD; kk += 32) {
    Frag a = load_frag(Qs, r0, kk, lane, LDT);
#pragma unroll
    for (int ct = 0; ct < 4; ++ct) {
      Frag bf = load_frag(Ks, ct * 16, kk, lane, LDT);
      sacc[ct] = wmma_bf16(a.v, bf.v, sacc[ct]);
    }
  }

  // ---- per-key max over this wave's 16 rows -> wmax[wave][key] ----
#pragma unroll
  for (int ct = 0; ct < 4; ++ct) {
    float m = sacc[ct][0];
#pragma unroll
    for (int r = 1; r < 8; ++r) m = fmaxf(m, sacc[ct][r]);
    m = fmaxf(m, __shfl_xor(m, 16, 32));      // combine rows r and r+8 (wave32)
    if (lane < 16) wmax[wave][ct * 16 + lane] = m;
  }
  __syncthreads();

  // ---- quirk: max over the 64-query block, then exp; stage W (bf16) ----
  const int blk = wave >> 2;                  // query block 0 (rows<64) or 1
  const int rbase = r0 + ((lane >> 4) << 3);
#pragma unroll
  for (int ct = 0; ct < 4; ++ct) {
    const int col = ct * 16 + (lane & 15);
    const float c = fmaxf(fmaxf(wmax[blk * 4 + 0][col], wmax[blk * 4 + 1][col]),
                          fmaxf(wmax[blk * 4 + 2][col], wmax[blk * 4 + 3][col]));
#pragma unroll
    for (int r = 0; r < 8; ++r) {
      const float w = __expf(sacc[ct][r] - c);
      Ws[(rbase + r) * LDT + col] = f32_to_bf16(w);
    }
  }
  __syncthreads();

  // ---- den[row] = 1e-10 + sum_t W[row][t]  (vectorized 8-bf16 reads) ----
  if (tid < 128) {
    float sum = 1e-10f;
    const u32x4* p = (const u32x4*)(Ws + tid * LDT);
#pragma unroll
    for (int i = 0; i < 8; ++i) {
      u32x4 d = p[i];
#pragma unroll
      for (int e = 0; e < 4; ++e) {
        sum += bf16_to_f32((u16)(d[e] & 0xFFFFu));
        sum += bf16_to_f32((u16)(d[e] >> 16));
      }
    }
    den[tid] = sum;
  }
  __syncthreads();

  // ---- num = W * V  (B operand from Vt = V^T, same N x K loader) ----
  v8f oacc[4];
#pragma unroll
  for (int ct = 0; ct < 4; ++ct) oacc[ct] = vzero;
#pragma unroll
  for (int kk = 0; kk < T; kk += 32) {
    Frag a = load_frag(Ws, r0, kk, lane, LDT);
#pragma unroll
    for (int ct = 0; ct < 4; ++ct) {
      Frag bf = load_frag(Vt, ct * 16, kk, lane, LDT);
      oacc[ct] = wmma_bf16(a.v, bf.v, oacc[ct]);
    }
  }

  float dinv[8];
#pragma unroll
  for (int r = 0; r < 8; ++r) dinv[r] = 1.0f / den[rbase + r];

  const size_t obase = ((size_t)(b * M) + s * SEG) * 1024 + h * 64;
#pragma unroll
  for (int ct = 0; ct < 4; ++ct) {
    const int col = ct * 16 + (lane & 15);
#pragma unroll
    for (int r = 0; r < 8; ++r)
      attn[obase + (size_t)(rbase + r) * 1024 + col] =
          f32_to_bf16(oacc[ct][r] * dinv[r]);
  }
}

// ---------------------------------------------------------------------------
extern "C" void kernel_launch(void* const* d_in, const int* in_sizes, int n_in,
                              void* d_out, int out_size, void* d_ws, size_t ws_size,
                              hipStream_t stream) {
  (void)in_sizes; (void)n_in; (void)out_size; (void)ws_size;
  const float* x    = (const float*)d_in[0];   // [4,4096,1024]
  const float* Wqkv = (const float*)d_in[1];   // [3072,1024]
  const float* Wout = (const float*)d_in[2];   // [1024,1024]
  const int*   hmap = (const int*)d_in[3];     // [4096]
  float* out = (float*)d_out;                  // [4,4096,1024]

  u16* qkv  = (u16*)d_ws;                            // bf16 [16384][3072] 96MB
  u16* attn = qkv + (size_t)16384 * 3072;            // bf16 [16384][1024] 32MB

  // K1: qkv = x @ Wqkv^T, q-component (cols < 1024) pre-scaled by 1/8.
  {
    dim3 grid(3072 / 128, 16384 / 128);
    gemm_bt_kernel<float, u16><<<grid, 256, 0, stream>>>(
        x, Wqkv, qkv, 3072, 1024, 1024);
  }
  // K2: segmented hilbert attention. grid = (S=32, H=16, B=4).
  {
    dim3 grid(32, 16, 4);
    hilbert_attn_kernel<<<grid, 256, 0, stream>>>(qkv, hmap, attn);
  }
  // K3: out = attn @ Wout^T.
  {
    dim3 grid(1024 / 128, 16384 / 128);
    gemm_bt_kernel<u16, float><<<grid, 256, 0, stream>>>(
        attn, Wout, out, 1024, 1024, 0);
  }
}
